// SparseAttentionBlock_80582176407602
// MI455X (gfx1250) — matrix operand order repaired and checked by reading, not compiled, not run on hardware
//
#include <hip/hip_runtime.h>
#include <hip/hip_bf16.h>

// ---------------- problem constants ----------------
#define B_    2
#define L_    4096
#define D_    512
#define H_    8
#define DH_   64
#define NTOK  (B_ * L_)          // 8192
#define DFF   2048

typedef __bf16 bf16_t;
typedef __bf16 v16bf __attribute__((ext_vector_type(16)));
typedef __bf16 v8bf  __attribute__((ext_vector_type(8)));
typedef float  v8f   __attribute__((ext_vector_type(8)));
typedef int    v4i   __attribute__((ext_vector_type(4)));

union V16U { v16bf v; v8bf h[2]; };

// Load a 16x32 A-operand / 32x16 B-operand half-pair (two 16-byte loads).
static __device__ inline v16bf load_ab(const bf16_t* p0, const bf16_t* p1) {
    V16U u;
    u.h[0] = *(const v8bf*)p0;
    u.h[1] = *(const v8bf*)p1;
    return u.v;
}

// ---------------- gfx1250 async global->LDS copy (ASYNCcnt path) ----------------
typedef __attribute__((address_space(1))) v4i glb_v4i;
typedef __attribute__((address_space(3))) v4i lds_v4i;

#if defined(__has_builtin)
#if __has_builtin(__builtin_amdgcn_global_load_async_to_lds_b128)
#define HAVE_ASYNC_LDS 1
#endif
#endif

static __device__ inline void copy16_lds(const bf16_t* g, bf16_t* l) {
#ifdef HAVE_ASYNC_LDS
    __builtin_amdgcn_global_load_async_to_lds_b128((glb_v4i*)g, (lds_v4i*)l, 0, 0);
#else
    *(v8bf*)l = *(const v8bf*)g;     // fallback: sync load + ds_store
#endif
}

#ifdef HAVE_ASYNC_LDS
#define WAIT_ASYNC() asm volatile("s_wait_asynccnt 0x0" ::: "memory")
#else
#define WAIT_ASYNC() do {} while (0)
#endif

// ---------------- fp32 -> bf16 weight conversion ----------------
__global__ void k_cvt_bf16(const float* __restrict__ in, bf16_t* __restrict__ out, int n) {
    int i = (blockIdx.x * blockDim.x + threadIdx.x) * 4;
    if (i + 3 < n) {
        float4 f = *(const float4*)(in + i);
        out[i + 0] = (bf16_t)f.x;
        out[i + 1] = (bf16_t)f.y;
        out[i + 2] = (bf16_t)f.z;
        out[i + 3] = (bf16_t)f.w;
    }
}

// ---------------- LayerNorm (fp32 in -> bf16 out), one row per block ----------------
__global__ void k_layernorm_bf16(const float* __restrict__ x, const float* __restrict__ g,
                                 const float* __restrict__ bta, bf16_t* __restrict__ out) {
    int row = blockIdx.x;
    const float* xr = x + (size_t)row * D_;
    int t = threadIdx.x;                       // 256 threads, 2 elems each
    float2 v = *(const float2*)(xr + t * 2);
    float s  = v.x + v.y;
    float s2 = v.x * v.x + v.y * v.y;
    #pragma unroll
    for (int m = 16; m >= 1; m >>= 1) {
        s  += __shfl_xor(s,  m, 32);
        s2 += __shfl_xor(s2, m, 32);
    }
    __shared__ float rs[8], rs2[8];
    int wave = t >> 5, lane = t & 31;
    if (lane == 0) { rs[wave] = s; rs2[wave] = s2; }
    __syncthreads();
    if (wave == 0) {
        float a  = (lane < 8) ? rs[lane]  : 0.f;
        float a2 = (lane < 8) ? rs2[lane] : 0.f;
        #pragma unroll
        for (int m = 4; m >= 1; m >>= 1) {
            a  += __shfl_xor(a,  m, 32);
            a2 += __shfl_xor(a2, m, 32);
        }
        if (lane == 0) { rs[0] = a; rs2[0] = a2; }
    }
    __syncthreads();
    float mu   = rs[0] * (1.f / D_);
    float var  = rs2[0] * (1.f / D_) - mu * mu;
    float rstd = rsqrtf(var + 1e-5f);
    int i0 = t * 2;
    out[(size_t)row * D_ + i0 + 0] = (bf16_t)((v.x - mu) * rstd * g[i0 + 0] + bta[i0 + 0]);
    out[(size_t)row * D_ + i0 + 1] = (bf16_t)((v.y - mu) * rstd * g[i0 + 1] + bta[i0 + 1]);
}

// =====================================================================
// WMMA GEMM core with LDS-staged, double-buffered weight panel.
// Block = 256 threads (8 waves): 128 tokens x 64 features.
// Weight panel per k-step: 64 feat x 32 k bf16 = 4KB, filled by one
// async b128 copy per thread (one ASYNCcnt op per wave), double buffered.
// =====================================================================
struct GemmTile {
    v8f acc[4];
    int tok0, m0, r, hf;
};

static __device__ inline void gemm_core(GemmTile& gt, const bf16_t* __restrict__ A,
                                        const bf16_t* __restrict__ W, int M, int K,
                                        bf16_t (*wbuf)[64 * 32]) {
    int tid  = threadIdx.x;
    int wave = tid >> 5, lane = tid & 31;
    gt.r  = lane & 15;
    gt.hf = lane >> 4;
    int mblocks = M / 64;
    int bn = blockIdx.x / mblocks;
    int bm = blockIdx.x % mblocks;
    gt.tok0 = bn * 128 + wave * 16;
    gt.m0   = bm * 64;
    #pragma unroll
    for (int t = 0; t < 4; t++) gt.acc[t] = v8f{};

    // async-copy mapping: thread -> (feature row, 16B segment)
    int crow = tid >> 2, cseg = tid & 3;
    const bf16_t* gsrc = W + (size_t)(gt.m0 + crow) * K + cseg * 8;
    bf16_t* ldst = &wbuf[0][crow * 32 + cseg * 8];   // [buf][row*32 + seg*8]

    // prologue: fill buffer 0 with k=0 panel
    copy16_lds(gsrc, ldst);

    const bf16_t* Arow = A + (size_t)(gt.tok0 + gt.r) * K;
    int nsteps = K / 32;
    for (int i = 0; i < nsteps; i++) {
        int k = i * 32;
        WAIT_ASYNC();                 // my panel-i copy done
        __syncthreads();              // everyone's panel-i writes visible; buf[i+1] free
        if (i + 1 < nsteps)
            copy16_lds(gsrc + k + 32, &wbuf[(i + 1) & 1][crow * 32 + cseg * 8]);
        if (k + 64 < K) __builtin_prefetch(Arow + k + 64, 0, 3);
        v16bf a = load_ab(Arow + k + gt.hf * 16, Arow + k + gt.hf * 16 + 8);
        const bf16_t* wb = wbuf[i & 1];
        #pragma unroll
        for (int t = 0; t < 4; t++) {
            const bf16_t* wrow = wb + (t * 16 + gt.r) * 32 + gt.hf * 16;
            v16bf b = load_ab(wrow, wrow + 8);
            gt.acc[t] = __builtin_amdgcn_wmma_f32_16x16x32_bf16(false, a, false, b,
                                                                (short)0, gt.acc[t], false, false);
        }
    }
}

// epi == 1: outF = resid + acc (fp32).  epi == 2: outB = bf16(silu(acc)).
__global__ void k_gemm_epi(const bf16_t* __restrict__ A, const bf16_t* __restrict__ W,
                           int M, int K,
                           const float* __restrict__ resid,
                           float* __restrict__ outF, bf16_t* __restrict__ outB, int epi) {
    __shared__ __align__(16) bf16_t wbuf[2][64 * 32];
    GemmTile gt;
    gemm_core(gt, A, W, M, K, wbuf);
    #pragma unroll
    for (int t = 0; t < 4; t++) {
        int col = gt.m0 + t * 16 + gt.r;
        #pragma unroll
        for (int v = 0; v < 8; v++) {
            int tok = gt.tok0 + gt.hf * 8 + v;
            float val = gt.acc[t][v];
            if (epi == 2) {
                outB[(size_t)tok * M + col] = (bf16_t)(val / (1.f + __expf(-val)));
            } else {
                outF[(size_t)tok * M + col] = resid[(size_t)tok * M + col] + val;
            }
        }
    }
}

// QKV GEMM with scatter into q[B,H,L,dh], k[B,H,L,dh], vT[B,H,dh,L]
__global__ void k_gemm_qkv(const bf16_t* __restrict__ A, const bf16_t* __restrict__ W,
                           bf16_t* __restrict__ q, bf16_t* __restrict__ kk,
                           bf16_t* __restrict__ vT) {
    __shared__ __align__(16) bf16_t wbuf[2][64 * 32];
    GemmTile gt;
    gemm_core(gt, A, W, 3 * D_, D_, wbuf);
    int type = gt.m0 / D_;                     // 0=q, 1=k, 2=v
    int h    = (gt.m0 % D_) / DH_;
    #pragma unroll
    for (int t = 0; t < 4; t++) {
        int d = t * 16 + gt.r;                 // 0..63 inside head
        #pragma unroll
        for (int v = 0; v < 8; v++) {
            int tok = gt.tok0 + gt.hf * 8 + v;
            int b   = tok / L_, l = tok % L_;
            size_t bh = (size_t)(b * H_ + h);
            bf16_t val = (bf16_t)gt.acc[t][v];
            if (type == 0)      q [(bh * L_ + l) * DH_ + d] = val;
            else if (type == 1) kk[(bh * L_ + l) * DH_ + d] = val;
            else                vT[(bh * DH_ + d) * L_ + l] = val;
        }
    }
}

// ---------------- flash attention: causal + temporal decay + key padding mask ----------------
// block = 256 threads (8 waves). Each wave owns a 16-row Q tile, streams 32-key blocks.
#define PS 40   // LDS row pitch (bf16 elems); 40*2=80B, 16B-aligned
__global__ void k_flash_attn(const bf16_t* __restrict__ q, const bf16_t* __restrict__ kk,
                             const bf16_t* __restrict__ vT, const float* __restrict__ ts,
                             const float* __restrict__ msk, const float* __restrict__ dr,
                             bf16_t* __restrict__ out /*[NTOK, D]*/) {
    __shared__ __align__(16) bf16_t plds[8][16 * PS];
    int tid  = threadIdx.x, wave = tid >> 5, lane = tid & 31;
    int r = lane & 15, hf = lane >> 4;
    int tilesPerBH = L_ / 128;                 // 32
    int bh = blockIdx.x / tilesPerBH;
    int qt = blockIdx.x % tilesPerBH;
    int b  = bh / H_, h = bh % H_;
    int qb = qt * 128 + wave * 16;

    const bf16_t* qbase = q  + (size_t)bh * L_ * DH_;
    const bf16_t* kbase = kk + (size_t)bh * L_ * DH_;
    const bf16_t* vbase = vT + (size_t)bh * DH_ * L_;
    const float*  tsb   = ts  + (size_t)b * L_;
    const float*  mb    = msk + (size_t)b * L_;
    float decay  = log1pf(__expf(dr[h]));      // softplus
    const float biasc = decay * (1.f / 24.f);
    const float scale = 0.125f;                // dh^-0.5
    const float NEG_INF = -__builtin_inff();

    // Q A-operands for the two dh chunks (loaded once)
    const bf16_t* qrow = qbase + (size_t)(qb + r) * DH_;
    v16bf aQ0 = load_ab(qrow +      hf * 16, qrow +      hf * 16 + 8);
    v16bf aQ1 = load_ab(qrow + 32 + hf * 16, qrow + 32 + hf * 16 + 8);

    float mrow[8], lrow[8], tq[8];
    v8f O[4] = {};
    #pragma unroll
    for (int v = 0; v < 8; v++) {
        mrow[v] = NEG_INF; lrow[v] = 0.f;
        tq[v] = tsb[qb + hf * 8 + v];
    }

    int kbMax = (qb + 15) >> 5;                // causal bound
    for (int kb = 0; kb <= kbMax; kb++) {
        int k0 = kb * 32;
        // ---- S = Q K^T for two 16-key sub-tiles ----
        v8f S[2] = {};
        #pragma unroll
        for (int t = 0; t < 2; t++) {
            const bf16_t* krow = kbase + (size_t)(k0 + t * 16 + r) * DH_;
            v16bf bK0 = load_ab(krow +      hf * 16, krow +      hf * 16 + 8);
            v16bf bK1 = load_ab(krow + 32 + hf * 16, krow + 32 + hf * 16 + 8);
            S[t] = __builtin_amdgcn_wmma_f32_16x16x32_bf16(false, aQ0, false, bK0,
                                                           (short)0, S[t], false, false);
            S[t] = __builtin_amdgcn_wmma_f32_16x16x32_bf16(false, aQ1, false, bK1,
                                                           (short)0, S[t], false, false);
        }
        // ---- bias + causal + padding mask (C layout: lane=column, VGPR=row) ----
        float p[2][8];
        #pragma unroll
        for (int t = 0; t < 2; t++) {
            int col = k0 + t * 16 + r;
            float tk = tsb[col];
            float mk = mb[col];
            #pragma unroll
            for (int v = 0; v < 8; v++) {
                int row = qb + hf * 8 + v;
                float s = S[t][v] * scale - biasc * fabsf(tq[v] - tk);
                if (col > row || mk == 0.f) s = NEG_INF;
                p[t][v] = s;
            }
        }
        // ---- online softmax (shfl over 16-lane half = the N dimension) ----
        float mnew[8], alpha[8];
        #pragma unroll
        for (int v = 0; v < 8; v++) {
            float mx = fmaxf(p[0][v], p[1][v]);
            mx = fmaxf(mx, __shfl_xor(mx, 1, 32));
            mx = fmaxf(mx, __shfl_xor(mx, 2, 32));
            mx = fmaxf(mx, __shfl_xor(mx, 4, 32));
            mx = fmaxf(mx, __shfl_xor(mx, 8, 32));
            mnew[v]  = fmaxf(mrow[v], mx);
            alpha[v] = __expf(mrow[v] - mnew[v]);
            mrow[v]  = mnew[v];
        }
        #pragma unroll
        for (int v = 0; v < 8; v++) {
            p[0][v] = __expf(p[0][v] - mnew[v]);
            p[1][v] = __expf(p[1][v] - mnew[v]);
            float sr = p[0][v] + p[1][v];
            sr += __shfl_xor(sr, 1, 32);
            sr += __shfl_xor(sr, 2, 32);
            sr += __shfl_xor(sr, 4, 32);
            sr += __shfl_xor(sr, 8, 32);
            lrow[v] = lrow[v] * alpha[v] + sr;
        }
        #pragma unroll
        for (int t = 0; t < 4; t++)
            #pragma unroll
            for (int v = 0; v < 8; v++) O[t][v] *= alpha[v];
        // ---- C-layout P -> LDS -> A-layout reload ----
        bf16_t* pl = plds[wave];
        #pragma unroll
        for (int t = 0; t < 2; t++)
            #pragma unroll
            for (int v = 0; v < 8; v++)
                pl[(hf * 8 + v) * PS + t * 16 + r] = (bf16_t)p[t][v];
        v16bf aP = load_ab(pl + r * PS + hf * 16, pl + r * PS + hf * 16 + 8);
        // ---- O += P * V  (V^T rows are key-contiguous) ----
        #pragma unroll
        for (int t = 0; t < 4; t++) {
            const bf16_t* vrow = vbase + (size_t)(t * 16 + r) * L_ + k0 + hf * 16;
            v16bf bV = load_ab(vrow, vrow + 8);
            O[t] = __builtin_amdgcn_wmma_f32_16x16x32_bf16(false, aP, false, bV,
                                                           (short)0, O[t], false, false);
        }
    }
    // ---- normalize and store attn output [NTOK, D] (bf16) ----
    #pragma unroll
    for (int v = 0; v < 8; v++) lrow[v] = 1.f / lrow[v];
    #pragma unroll
    for (int t = 0; t < 4; t++) {
        int d = t * 16 + r;
        #pragma unroll
        for (int v = 0; v < 8; v++) {
            int l   = qb + hf * 8 + v;
            int tok = b * L_ + l;
            out[(size_t)tok * D_ + h * DH_ + d] = (bf16_t)(O[t][v] * lrow[v]);
        }
    }
}

// ---------------- launcher ----------------
extern "C" void kernel_launch(void* const* d_in, const int* in_sizes, int n_in,
                              void* d_out, int out_size, void* d_ws, size_t ws_size,
                              hipStream_t stream) {
    const float* x    = (const float*)d_in[0];
    const float* ts   = (const float*)d_in[1];
    const float* msk  = (const float*)d_in[2];
    const float* g1   = (const float*)d_in[3];
    const float* b1   = (const float*)d_in[4];
    const float* wqkv = (const float*)d_in[5];
    const float* wout = (const float*)d_in[6];
    const float* dr   = (const float*)d_in[7];
    const float* g2   = (const float*)d_in[8];
    const float* b2   = (const float*)d_in[9];
    const float* wff1 = (const float*)d_in[10];
    const float* wff2 = (const float*)d_in[11];

    char* ws = (char*)d_ws;
    size_t off = 0;
    auto alloc = [&](size_t bytes) -> char* {
        char* p = ws + off;
        off += (bytes + 255) & ~(size_t)255;
        return p;
    };
    bf16_t* Wqkv = (bf16_t*)alloc((size_t)3 * D_ * D_ * 2);
    bf16_t* Wout = (bf16_t*)alloc((size_t)D_ * D_ * 2);
    bf16_t* Wff1 = (bf16_t*)alloc((size_t)DFF * D_ * 2);
    bf16_t* Wff2 = (bf16_t*)alloc((size_t)D_ * DFF * 2);
    bf16_t* XN   = (bf16_t*)alloc((size_t)NTOK * D_ * 2);
    bf16_t* Qb   = (bf16_t*)alloc((size_t)NTOK * D_ * 2);
    bf16_t* Kb   = (bf16_t*)alloc((size_t)NTOK * D_ * 2);
    bf16_t* VT   = (bf16_t*)alloc((size_t)NTOK * D_ * 2);
    bf16_t* AO   = (bf16_t*)alloc((size_t)NTOK * D_ * 2);
    float*  X2   = (float*) alloc((size_t)NTOK * D_ * 4);
    bf16_t* Hb   = (bf16_t*)alloc((size_t)NTOK * D_ * 2);
    bf16_t* F1   = (bf16_t*)alloc((size_t)NTOK * DFF * 2);

    // weight conversions (4 elems / thread)
    k_cvt_bf16<<<(3 * D_ * D_) / 1024, 256, 0, stream>>>(wqkv, Wqkv, 3 * D_ * D_);
    k_cvt_bf16<<<(D_ * D_) / 1024, 256, 0, stream>>>(wout, Wout, D_ * D_);
    k_cvt_bf16<<<(DFF * D_) / 1024, 256, 0, stream>>>(wff1, Wff1, DFF * D_);
    k_cvt_bf16<<<(D_ * DFF) / 1024, 256, 0, stream>>>(wff2, Wff2, D_ * DFF);

    // LN1
    k_layernorm_bf16<<<NTOK, 256, 0, stream>>>(x, g1, b1, XN);

    // QKV projection + scatter: grid = (NTOK/128) * (1536/64) = 64*24
    k_gemm_qkv<<<64 * 24, 256, 0, stream>>>(XN, Wqkv, Qb, Kb, VT);

    // flash attention (B*H*L/128 = 512 blocks)
    k_flash_attn<<<B_ * H_ * (L_ / 128), 256, 0, stream>>>(Qb, Kb, VT, ts, msk, dr, AO);

    // out-proj + residual: X2 = x + AO @ Wout^T   (64*8 blocks)
    k_gemm_epi<<<64 * 8, 256, 0, stream>>>(AO, Wout, D_, D_, x, X2, nullptr, 1);

    // LN2
    k_layernorm_bf16<<<NTOK, 256, 0, stream>>>(X2, g2, b2, Hb);

    // FFN1 + SiLU: F1 = silu(Hb @ Wff1^T)   (64*32 blocks)
    k_gemm_epi<<<64 * 32, 256, 0, stream>>>(Hb, Wff1, DFF, D_, nullptr, nullptr, F1, 2);

    // FFN2 + residual: out = X2 + F1 @ Wff2^T   (64*8 blocks)
    k_gemm_epi<<<64 * 8, 256, 0, stream>>>(F1, Wff2, D_, DFF, X2, (float*)d_out, nullptr, 1);
}
